// KANLayer_65919158059310
// MI455X (gfx1250) — matrix operand order repaired and checked by reading, not compiled, not run on hardware
//
#include <hip/hip_runtime.h>

// ---------------------------------------------------------------------------
// KAN layer as ONE bf16 WMMA GEMM on gfx1250 (wave32).
//   out[b,o] = sum_i silu(x[b,i]) * (imp*Wb)[i,o]
//            + sum_{i,c} basis_c(x[b,i]) * (cp[i,o,c]*imp*Ws[i,o])
// K layout: k = i*6 + c   (c=0: silu column, c=1..5: basis columns)
// ---------------------------------------------------------------------------

typedef __bf16 bf16_t;
typedef bf16_t v16bf __attribute__((ext_vector_type(16)));
typedef bf16_t v8bf  __attribute__((ext_vector_type(8)));
typedef float  v8f   __attribute__((ext_vector_type(8)));

#define B_DIM 8192
#define I_DIM 512
#define O_DIM 512
#define NCP   5
#define KEXP  6                 // 1 silu + 5 basis per input channel
#define K_DIM (I_DIM * KEXP)    // 3072

#define BM  128
#define BN  128
#define BK  96                  // 16 input channels * 6
#define IK  16                  // input channels per K tile
#define LDT 104                 // padded LDS row stride in bf16 (52 dwords -> conflict-free)
#define NT  512                 // threads per block (16 waves)

static __device__ __forceinline__ unsigned short f2bf(float f) {
  union { float f; unsigned u; } v; v.f = f;
  unsigned r = ((v.u >> 16) & 1u) + 0x7fffu;   // round-to-nearest-even
  return (unsigned short)((v.u + r) >> 16);
}

// ---------------------------------------------------------------------------
// Prep: W^T [O][K] in bf16, k = i*6 + c.
// ---------------------------------------------------------------------------
__global__ __launch_bounds__(256) void kan_prep_w(
    const float* __restrict__ cp,  const float* __restrict__ bw,
    const float* __restrict__ sw,  const float* __restrict__ imp,
    unsigned short* __restrict__ wt) {
  int t = blockIdx.x * 256 + threadIdx.x;   // 0 .. I*O-1
  int i = t & (I_DIM - 1);
  int o = t >> 9;
  int io = i * O_DIM + o;
  float im = imp[io];
  float wbase = im * bw[io];
  float wspl  = im * sw[io];
  unsigned short h[6];
  h[0] = f2bf(wbase);
#pragma unroll
  for (int c = 0; c < NCP; ++c) h[1 + c] = f2bf(cp[(size_t)io * NCP + c] * wspl);
  unsigned* dst = (unsigned*)(wt + (size_t)o * K_DIM + i * KEXP);  // 4B aligned
  dst[0] = (unsigned)h[0] | ((unsigned)h[1] << 16);
  dst[1] = (unsigned)h[2] | ((unsigned)h[3] << 16);
  dst[2] = (unsigned)h[4] | ((unsigned)h[5] << 16);
}

// ---------------------------------------------------------------------------
// Fused expand + GEMM. 512 threads = 16 waves; wave grid 4(M) x 4(N);
// each wave: 2x2 tiles of v_wmma_f32_16x16x32_bf16 over a 32x32 region.
// ---------------------------------------------------------------------------
__global__ __launch_bounds__(NT) void kan_wmma_gemm(
    const float* __restrict__ x, const unsigned short* __restrict__ wt,
    float* __restrict__ out) {
  __shared__ unsigned short Asm[BM * LDT];
  __shared__ unsigned short Bsm[BN * LDT];

  const int tid  = threadIdx.x;
  const int lane = tid & 31;
  const int wid  = tid >> 5;        // 0..15
  const int m0 = blockIdx.x * BM;
  const int n0 = blockIdx.y * BN;
  const int wm = (wid >> 2) * 32;   // wave M offset within tile
  const int wn = (wid & 3) * 32;    // wave N offset within tile

  // Normalized Gaussian pseudo-B-spline basis at grid points {-1, 0, +1}
  // (compile-time constants; consumed as fma literals below)
  constexpr float R0[5] = {0.543362f, 0.317131f, 0.112264f, 0.024104f, 0.003139f};
  constexpr float R1[5] = {0.083905f, 0.228077f, 0.376036f, 0.228077f, 0.083905f};
  constexpr float R2[5] = {0.003139f, 0.024104f, 0.112264f, 0.317131f, 0.543362f};

  v8f acc[2][2];
#pragma unroll
  for (int a = 0; a < 2; ++a)
#pragma unroll
    for (int b = 0; b < 2; ++b) acc[a][b] = (v8f)0.0f;

  for (int kt = 0; kt < K_DIM / BK; ++kt) {
    const int i0 = kt * IK;

    // ---- A tile: expand x[128 x 16] -> bf16 [128 x 96] in LDS ----
#pragma unroll
    for (int rep = 0; rep < 4; ++rep) {
      int e  = tid + rep * NT;         // 0..2047
      int m  = e >> 4;
      int ii = e & 15;
      float xv  = x[(size_t)(m0 + m) * I_DIM + i0 + ii];
      float sig = 1.0f / (1.0f + __expf(-xv));
      float silu = xv * sig;
      float xc = fminf(1.0f, fmaxf(-1.0f, xv));
      float gi = xc + 1.0f;            // grid index in [0,2]
      int lo = (int)gi;                // floor (gi >= 0); lo in {0,1,2}
      if (lo > 2) lo = 2;
      float frac = gi - (float)lo;     // 0 at integer grid points
      // Row weights: basis[c] = w0*R0[c] + w1*R1[c] + w2*R2[c]
      // (pure cndmask + fma with literal constants; no table in memory)
      float wlo = 1.0f - frac;
      float w0 = (lo == 0) ? wlo  : 0.0f;
      float w1 = (lo == 0) ? frac : ((lo == 1) ? wlo : 0.0f);
      float w2 = (lo == 1) ? frac : ((lo == 2) ? wlo : 0.0f);
      unsigned short h[6];
      h[0] = f2bf(silu);
#pragma unroll
      for (int c = 0; c < 5; ++c)
        h[1 + c] = f2bf(w0 * R0[c] + w1 * R1[c] + w2 * R2[c]);
      unsigned* dst = (unsigned*)&Asm[m * LDT + ii * KEXP];  // 12*ii bytes -> 4B aligned
      dst[0] = (unsigned)h[0] | ((unsigned)h[1] << 16);
      dst[1] = (unsigned)h[2] | ((unsigned)h[3] << 16);
      dst[2] = (unsigned)h[4] | ((unsigned)h[5] << 16);
    }

    // ---- B tile: W^T [128 n-rows x 96 k] bf16, 16B vector copies ----
    const unsigned short* wsrc = wt + (size_t)n0 * K_DIM + kt * BK;
#pragma unroll
    for (int rep = 0; rep < 3; ++rep) {
      int ch = tid + rep * NT;         // 0..1535, each chunk = 8 bf16
      int n  = ch / 12;
      int kc = (ch % 12) * 8;
      uint4 v = *(const uint4*)(wsrc + (size_t)n * K_DIM + kc);
      *(uint4*)&Bsm[n * LDT + kc] = v;
      if (kt + 1 < K_DIM / BK)         // warm L2/L0 for the next K tile
        __builtin_prefetch(wsrc + BK + (size_t)n * K_DIM + kc, 0, 1);
    }
    __syncthreads();

    // ---- 3 x (4 WMMA) per K tile ----
#pragma unroll
    for (int kk = 0; kk < BK; kk += 32) {
      v16bf afrag[2];
#pragma unroll
      for (int a = 0; a < 2; ++a) {
        int row = wm + a * 16 + (lane & 15);
        int hb  = (lane >> 4) << 3;    // lanes 16-31: +8 in K
        v8bf lo8 = *(const v8bf*)&Asm[row * LDT + kk + hb];
        v8bf hi8 = *(const v8bf*)&Asm[row * LDT + kk + 16 + hb];
        afrag[a] = __builtin_shufflevector(lo8, hi8, 0,1,2,3,4,5,6,7,8,9,10,11,12,13,14,15);
      }
      v16bf bfrag[2];
#pragma unroll
      for (int b = 0; b < 2; ++b) {
        int col = wn + b * 16 + (lane & 15);
        int kb  = kk + ((lane >> 4) << 3);  // lanes 16-31: K+8
        v8bf lo8 = *(const v8bf*)&Bsm[col * LDT + kb];
        v8bf hi8 = *(const v8bf*)&Bsm[col * LDT + kb + 16];
        bfrag[b] = __builtin_shufflevector(lo8, hi8, 0,1,2,3,4,5,6,7,8,9,10,11,12,13,14,15);
      }
#pragma unroll
      for (int a = 0; a < 2; ++a)
#pragma unroll
        for (int b = 0; b < 2; ++b)
          acc[a][b] = __builtin_amdgcn_wmma_f32_16x16x32_bf16(
              false, afrag[a], false, bfrag[b], (short)0, acc[a][b], false, false);
    }
    __syncthreads();
  }

  // ---- store C: lanes 0-15 -> M=r, lanes 16-31 -> M=8+r; N = lane%16 ----
#pragma unroll
  for (int a = 0; a < 2; ++a)
#pragma unroll
    for (int b = 0; b < 2; ++b) {
      int col   = n0 + wn + b * 16 + (lane & 15);
      int rbase = m0 + wm + a * 16 + ((lane & 16) ? 8 : 0);
#pragma unroll
      for (int r = 0; r < 8; ++r)
        out[(size_t)(rbase + r) * O_DIM + col] = acc[a][b][r];
    }
}

// ---------------------------------------------------------------------------
extern "C" void kernel_launch(void* const* d_in, const int* in_sizes, int n_in,
                              void* d_out, int out_size, void* d_ws, size_t ws_size,
                              hipStream_t stream) {
  const float* x   = (const float*)d_in[0];
  const float* cp  = (const float*)d_in[1];
  const float* bw  = (const float*)d_in[2];
  const float* sw  = (const float*)d_in[3];
  const float* imp = (const float*)d_in[4];
  unsigned short* wt = (unsigned short*)d_ws;   // needs K_DIM*O_DIM*2 = 3 MiB scratch

  kan_prep_w<<<(I_DIM * O_DIM) / 256, 256, 0, stream>>>(cp, bw, sw, imp, wt);

  dim3 grid(B_DIM / BM, O_DIM / BN);            // 64 x 4 = 256 workgroups
  kan_wmma_gemm<<<grid, NT, 0, stream>>>(x, wt, (float*)d_out);
}
